// DYANEnc_23244363006307
// MI455X (gfx1250) — compile-verified, matrix operand order checked by reading, not run on hardware
//
#include <hip/hip_runtime.h>
#include <hip/hip_bf16.h>
#include <math.h>

// ---- problem constants ----
#define B_   1024
#define T_   36
#define M_   150
#define K_   161          // 2*80+1 dictionary atoms
#define KP   176          // K rows padded to 11*16 (output tiles)
#define JP   224          // inner dim: 161 (y) + 36 (Y^T) + pad  = 7*32
#define MP   160          // m padded to 10*16
#define LSTR 232          // LDS row stride (bf16 elems), conflict padding
#define NTN  10
#define NTILES 110        // 11 * 10 output tiles of 16x16
#define LAM_ 0.1f
#define THR_ 1e-5f
#define MAXIT 100

#define C_CNT  (B_*K_*M_)           // 24,729,600
#define D_OFF  (C_CNT)
#define R_OFF  (C_CNT + T_*K_)      // 24,735,396

typedef __attribute__((ext_vector_type(16))) __bf16 v16bf;
typedef __attribute__((ext_vector_type(8)))  float  v8f;
typedef __attribute__((ext_vector_type(4)))  int    v4i;

union Frag  { uint4 q[2]; v16bf v; };
union YPack { unsigned short s[8]; uint4 q; };

#if __has_builtin(__builtin_amdgcn_global_load_async_to_lds_b128)
#define HAVE_ASYNC_LDS 1
#else
#define HAVE_ASYNC_LDS 0
#endif

__device__ __forceinline__ void stage16(const unsigned short* g, unsigned short* l) {
#if HAVE_ASYNC_LDS
  __builtin_amdgcn_global_load_async_to_lds_b128(
      (__attribute__((address_space(1))) v4i*)g,
      (__attribute__((address_space(3))) v4i*)l, 0, 0);
#else
  *(uint4*)l = *(const uint4*)g;
#endif
}

__device__ __forceinline__ void stage_wait() {
#if HAVE_ASYNC_LDS
#if __has_builtin(__builtin_amdgcn_s_wait_asynccnt)
  __builtin_amdgcn_s_wait_asynccnt(0);
#else
  asm volatile("s_wait_asynccnt 0x0" ::: "memory");
#endif
#endif
}

__device__ __forceinline__ unsigned short f2bf(float f) {
  unsigned u = __float_as_uint(f);
  unsigned r = u + 0x7FFFu + ((u >> 16) & 1u);   // round-to-nearest-even
  return (unsigned short)(r >> 16);
}

// ---- kernel 1: build dictionary D[t][k] ----
__global__ void build_D(const float* __restrict__ Drr, const float* __restrict__ Dth,
                        float* __restrict__ Dw, float* __restrict__ out) {
  int idx = blockIdx.x * blockDim.x + threadIdx.x;
  if (idx >= T_ * K_) return;
  int t = idx / K_, k = idx % K_;
  float v;
  if (k == 0) v = 1.0f;
  else if (k <= 80) { float rr = Drr[k-1], th = Dth[k-1];
                      v = powf(rr, (float)t) * cosf((float)t * th); }
  else              { float rr = Drr[k-81], th = Dth[k-81];
                      v = powf(rr, (float)t) * sinf((float)t * th); }
  Dw[idx] = v;
  out[D_OFF + idx] = v;
}

// ---- kernel 2: DtD, L = ||DtD||_F, A' = [I - DtD/L | (1/L) D^T] in bf16 ----
__global__ __launch_bounds__(256)
void prep(const float* __restrict__ Dw, unsigned short* __restrict__ Abf,
          float* __restrict__ scal, float* __restrict__ diff, int* __restrict__ done) {
  extern __shared__ char sm[];
  float* DtD = (float*)sm;                       // 161*161
  float* red = (float*)(sm + K_*K_*4);           // 256
  int tid = threadIdx.x;

  float part = 0.f;
  for (int e = tid; e < K_*K_; e += 256) {
    int i = e / K_, j = e % K_;
    float s = 0.f;
    for (int t = 0; t < T_; ++t) s = fmaf(Dw[t*K_ + i], Dw[t*K_ + j], s);
    DtD[e] = s;
    part += s * s;
  }
  red[tid] = part;
  __syncthreads();
  for (int s = 128; s > 0; s >>= 1) { if (tid < s) red[tid] += red[tid+s]; __syncthreads(); }
  float linv = 1.0f / sqrtf(red[0]);
  if (tid == 0) { scal[0] = linv; scal[1] = LAM_ * linv; *done = 0; }
  if (tid < MAXIT) diff[tid] = 0.f;
  __syncthreads();

  for (int e = tid; e < KP*JP; e += 256) {
    int k = e / JP, j = e % JP;
    float v = 0.f;
    if (k < K_) {
      if (j < K_)         v = (k == j ? 1.f : 0.f) - DtD[k*K_ + j] * linv;
      else if (j < K_+T_) v = linv * Dw[(j - K_)*K_ + k];
    }
    Abf[e] = f2bf(v);
  }
}

// ---- kernel 3: init y buffer [b][m][224]: cols 0..160 = y=0, 161..196 = bf16(Y^T) ----
__global__ void init_y(const float* __restrict__ Y, unsigned short* __restrict__ yg) {
  size_t idx = (size_t)blockIdx.x * blockDim.x + threadIdx.x;
  if (idx >= (size_t)B_ * MP * JP) return;
  int b = (int)(idx / (MP*JP));
  int r = (int)(idx % (MP*JP));
  int m = r / JP, j = r % JP;
  unsigned short v = 0;
  if (j >= K_ && j < K_+T_ && m < M_)
    v = f2bf(Y[(size_t)b*T_*M_ + (size_t)(j - K_)*M_ + m]);
  yg[idx] = v;
}

// ---- kernel 4: zero x (= output C region) ----
__global__ void zero_x(float* __restrict__ x) {
  size_t idx = (size_t)blockIdx.x * blockDim.x + threadIdx.x;
  if (idx < (size_t)C_CNT) x[idx] = 0.f;
}

// ---- kernel 5: one FISTA step (the WMMA engine) ----
__global__ __launch_bounds__(256)
void fista_step(const unsigned short* __restrict__ Abf, unsigned short* __restrict__ yg,
                float* __restrict__ x, const float* __restrict__ scal,
                float* __restrict__ diff, const int* __restrict__ done,
                float tt, int iter) {
  if (*done) return;                                    // frozen: x,y unchanged
  extern __shared__ char smem[];
  unsigned short* sA = (unsigned short*)smem;                      // [176][232]
  unsigned short* sY = (unsigned short*)(smem + KP*LSTR*2);        // [160][232]
  float* sRed = (float*)(smem + (KP+MP)*LSTR*2);                   // [256]

  int tid = threadIdx.x;
  int b = blockIdx.x;
  const unsigned short* ygb = yg + (size_t)b * MP * JP;
  float lam_eff = scal[1];

  // stage A' and y_b^T into LDS (b128 chunks; async-to-LDS when available)
  for (int c = tid; c < (KP*JP)/8; c += 256) {
    int row = c / (JP/8), jc = c % (JP/8);
    stage16(Abf + row*JP + jc*8, sA + row*LSTR + jc*8);
  }
  for (int c = tid; c < (MP*JP)/8; c += 256) {
    int row = c / (JP/8), jc = c % (JP/8);
    stage16(ygb + (size_t)row*JP + jc*8, sY + row*LSTR + jc*8);
  }
  stage_wait();
  __syncthreads();

  int wave = tid >> 5, lane = tid & 31;
  int half = (lane >> 4) & 1, lrow = lane & 15;
  float dsq = 0.f;

  for (int tile = wave; tile < NTILES; tile += 8) {
    int tm = tile / NTN, tn = tile % NTN;
    int m = tn*16 + lrow;
    int kbase = tm*16 + half*8;
    const unsigned short* ar = sA + (tm*16 + lrow)*LSTR + half*8;  // A row M
    const unsigned short* br = sY + (tn*16 + lrow)*LSTR + half*8;  // y^T row = col N
    bool mok   = (m < M_);
    bool fullk = (tm < 10);                 // all 8 k indices valid
    float* xrow = x + (size_t)b*(K_*M_) + m;

    // batched x_old prefetch, issued ahead of the WMMA chain to hide latency
    float xo[8];
    if (mok) {
      if (fullk) {
#pragma unroll
        for (int r = 0; r < 8; ++r) xo[r] = xrow[(size_t)(kbase + r)*M_];
      } else if (half == 0) {
        xo[0] = xrow[(size_t)160*M_];       // only k=160 valid in last tile row
      }
    }

    v8f acc = (v8f)0.0f;
#pragma unroll
    for (int kk = 0; kk < 7; ++kk) {
      Frag fa, fb;
      fa.q[0] = *(const uint4*)(ar + kk*32);
      fa.q[1] = *(const uint4*)(ar + kk*32 + 16);
      fb.q[0] = *(const uint4*)(br + kk*32);
      fb.q[1] = *(const uint4*)(br + kk*32 + 16);
      acc = __builtin_amdgcn_wmma_f32_16x16x32_bf16(
                false, fa.v, false, fb.v, (short)0, acc, false, false);
    }

    if (mok) {
      if (fullk) {
        YPack yp;
#pragma unroll
        for (int r = 0; r < 8; ++r) {
          float ay  = acc[r];                           // A*y + DtY (folded)
          float mag = fabsf(ay) - lam_eff;
          float xn  = mag > 0.f ? copysignf(mag, ay) : 0.f;
          float d   = xn - xo[r];
          dsq += d * d;
          xrow[(size_t)(kbase + r)*M_] = xn;
          yp.s[r] = f2bf((1.f + tt)*xn - tt*xo[r]);
        }
        // 8 consecutive k -> one 16B store into y^T row m
        *(uint4*)(yg + (size_t)b*MP*JP + (size_t)m*JP + kbase) = yp.q;
      } else if (half == 0) {
        float ay  = acc[0];
        float mag = fabsf(ay) - lam_eff;
        float xn  = mag > 0.f ? copysignf(mag, ay) : 0.f;
        float d   = xn - xo[0];
        dsq += d * d;
        xrow[(size_t)160*M_] = xn;
        yg[(size_t)b*MP*JP + (size_t)m*JP + 160] = f2bf((1.f + tt)*xn - tt*xo[0]);
      }
    }
  }

  sRed[tid] = dsq;
  __syncthreads();
  for (int s = 128; s > 0; s >>= 1) { if (tid < s) sRed[tid] += sRed[tid+s]; __syncthreads(); }
  if (tid == 0) atomicAdd(&diff[iter], sRed[0]);
}

// ---- kernel 6: early-exit check ----
__global__ void fista_fin(const float* __restrict__ diff, int* __restrict__ done, int iter) {
  if (!*done && sqrtf(diff[iter]) * (1.0f / (float)K_) < THR_) *done = 1;
}

// ---- kernel 7: reconstruction R = D @ C ----
__global__ void recon(const float* __restrict__ Dw, const float* __restrict__ x,
                      float* __restrict__ out) {
  size_t idx = (size_t)blockIdx.x * blockDim.x + threadIdx.x;
  if (idx >= (size_t)B_ * T_ * M_) return;
  int b = (int)(idx / (T_*M_));
  int r = (int)(idx % (T_*M_));
  int t = r / M_, m = r % M_;
  const float* Dt = Dw + t*K_;
  const float* xb = x + (size_t)b*K_*M_ + m;
  float s = 0.f;
  for (int k = 0; k < K_; ++k) s = fmaf(Dt[k], xb[(size_t)k*M_], s);
  out[R_OFF + idx] = s;
}

extern "C" void kernel_launch(void* const* d_in, const int* in_sizes, int n_in,
                              void* d_out, int out_size, void* d_ws, size_t ws_size,
                              hipStream_t stream) {
  const float* Y   = (const float*)d_in[0];
  const float* Drr = (const float*)d_in[1];
  const float* Dth = (const float*)d_in[2];
  float* out = (float*)d_out;
  float* x   = out;                                 // C region doubles as x

  char* ws = (char*)d_ws;
  unsigned short* Abf = (unsigned short*)ws;                        // 176*224 bf16
  unsigned short* yg  = (unsigned short*)(ws + (size_t)KP*JP*2);    // [B][160][224] bf16
  float* Dw   = (float*)(ws + (size_t)KP*JP*2 + (size_t)B_*MP*JP*2);
  float* scal = Dw + T_*K_;                                         // {linv, lam_eff}
  float* diff = scal + 2;                                           // [100]
  int*   done = (int*)(diff + MAXIT);

  // deterministic FISTA momentum sequence
  float tts[MAXIT]; float t = 1.f;
  for (int i = 0; i < MAXIT; ++i) {
    float tn = (1.f + sqrtf(1.f + 4.f*t*t)) * 0.5f;
    tts[i] = (t - 1.f) / tn; t = tn;
  }

  build_D<<<(T_*K_ + 255)/256, 256, 0, stream>>>(Drr, Dth, Dw, out);
  prep<<<1, 256, K_*K_*4 + 256*4, stream>>>(Dw, Abf, scal, diff, done);

  size_t yelems = (size_t)B_*MP*JP;
  init_y<<<(unsigned)((yelems + 255)/256), 256, 0, stream>>>(Y, yg);
  zero_x<<<(C_CNT + 255)/256, 256, 0, stream>>>(x);

  size_t stepLds = (size_t)(KP + MP)*LSTR*2 + 256*4;                // 156,928 B
  for (int i = 0; i < MAXIT; ++i) {
    fista_step<<<B_, 256, stepLds, stream>>>(Abf, yg, x, scal, diff, done, tts[i], i);
    fista_fin<<<1, 1, 0, stream>>>(diff, done, i);
  }

  recon<<<(unsigned)(((size_t)B_*T_*M_ + 255)/256), 256, 0, stream>>>(Dw, x, out);
}
